// SimpleTernaryNet_66211215835464
// MI455X (gfx1250) — compile-verified, matrix operand order checked by reading, not run on hardware
//
#include <hip/hip_runtime.h>
#include <hip/hip_bf16.h>

typedef __attribute__((ext_vector_type(16))) _Float16 v16h;
typedef __attribute__((ext_vector_type(8)))  _Float16 v8h;
typedef __attribute__((ext_vector_type(8)))  float    v8f;
typedef _Float16 half_t;

// ---------------- WMMA helpers ----------------
__device__ inline v8f wmma_f16(v16h a, v16h b, v8f c) {
  // D = A(16x32 f16) * B(32x16 f16) + C(16x16 f32)
  return __builtin_amdgcn_wmma_f32_16x16x32_f16(
      /*neg_a=*/false, a, /*neg_b=*/false, b,
      /*c_mod=*/(short)0, c, /*reuse_a=*/false, /*reuse_b=*/false);
}

__device__ inline v16h pack16(v8h lo, v8h hi) {
  v16h r;
#pragma unroll
  for (int i = 0; i < 8; ++i) { r[i] = lo[i]; r[8 + i] = hi[i]; }
  return r;
}

__device__ inline float block_reduce_sum(float v, float* red) {
  int t = threadIdx.x;
  red[t] = v;
  __syncthreads();
  for (int off = 128; off; off >>= 1) {
    if (t < off) red[t] += red[t + off];
    __syncthreads();
  }
  float r = red[0];
  __syncthreads();
  return r;
}

// ---------------- Workspace layout (bytes) ----------------
// B matrices stored column-major as [n][Kpad] f16 (alpha-scaled ternary).
static constexpr size_t OFF_B1  = 0;                     // 16  x 32   f16 -> 1 KB
static constexpr size_t OFF_B2  = 1024;                  // 32  x 160  f16 -> 10 KB
static constexpr size_t OFF_BF1 = 11264;                 // 128 x 2048 f16 -> 512 KB
static constexpr size_t OFF_BF2 = 535552;                // 16  x 128  f16 -> 4 KB
static constexpr size_t OFF_H1  = 540672;                // B x16x16x16 NHWC f16 -> 32 MB
static constexpr size_t OFF_H2  = OFF_H1 + (size_t)4096 * 16 * 16 * 16 * 2;  // B x8x8x32 NHWC f16
static constexpr size_t OFF_H3  = OFF_H2 + (size_t)4096 * 8 * 8 * 32 * 2;    // B x128 f16

// ============================================================
// Kernel 0: ternarize all 4 weight tensors, emit packed f16 B
// matrices (col-major [n][Kpad]) with the K ordering each
// consumer GEMM expects. 4 blocks x 256 threads.
// ============================================================
__global__ __launch_bounds__(256)
void tern_pack_kernel(const float* __restrict__ w1, const float* __restrict__ w2,
                      const float* __restrict__ wf1, const float* __restrict__ wf2,
                      half_t* __restrict__ B1, half_t* __restrict__ B2,
                      half_t* __restrict__ Bf1, half_t* __restrict__ Bf2) {
  __shared__ float red[256];
  int t = threadIdx.x;
  const float* w; int n;
  switch (blockIdx.x) {
    case 0:  w = w1;  n = 16 * 27;    break;
    case 1:  w = w2;  n = 32 * 144;   break;
    case 2:  w = wf1; n = 128 * 2048; break;
    default: w = wf2; n = 10 * 128;   break;
  }
  float s = 0.f;
  for (int i = t; i < n; i += 256) s += fabsf(w[i]);
  float delta = 0.7f * block_reduce_sum(s, red) / (float)n;
  float sm = 0.f, cnt = 0.f;
  for (int i = t; i < n; i += 256) {
    float a = fabsf(w[i]);
    if (a > delta) { sm += a; cnt += 1.f; }
  }
  float smt  = block_reduce_sum(sm, red);
  float cntt = block_reduce_sum(cnt, red);
  float alpha = smt / fmaxf(cntt, 1.f);

  auto tern = [&](float x) -> float {
    return (fabsf(x) > delta) ? (x > 0.f ? alpha : -alpha) : 0.f;
  };

  if (blockIdx.x == 0) {
    // B1[n][k], Kpad=32, k = ic*9+ky*3+kx (natural OIHW flatten)
    for (int idx = t; idx < 16 * 32; idx += 256) {
      int no = idx >> 5, k = idx & 31;
      float v = (k < 27) ? tern(w1[no * 27 + k]) : 0.f;
      B1[no * 32 + k] = (half_t)v;
    }
  } else if (blockIdx.x == 1) {
    // B2[n][k'], Kpad=160, k' = (ky*3+kx)*16 + ic  (NHWC-friendly)
    for (int idx = t; idx < 32 * 160; idx += 256) {
      int no = idx / 160, kp = idx % 160;
      float v = 0.f;
      if (kp < 144) {
        int sidx = kp >> 4, ic = kp & 15;
        int ky = sidx / 3, kx = sidx % 3;
        v = tern(w2[((no * 16 + ic) * 3 + ky) * 3 + kx]);
      }
      B2[no * 160 + kp] = (half_t)v;
    }
  } else if (blockIdx.x == 2) {
    // Bf1[n][k'], k' = (y*8+x)*32 + c  matches NHWC-flat h2;
    // reference k = c*64 + y*8 + x (NCHW flatten).
    for (int idx = t; idx < 128 * 2048; idx += 256) {
      int no = idx >> 11, kp = idx & 2047;
      int c = kp & 31, xx = (kp >> 5) & 7, yy = kp >> 8;
      int k = c * 64 + yy * 8 + xx;
      Bf1[no * 2048 + kp] = (half_t)tern(wf1[no * 2048 + k]);
    }
  } else {
    // Bf2[n][k], n padded 10->16
    for (int idx = t; idx < 16 * 128; idx += 256) {
      int no = idx >> 7, k = idx & 127;
      float v = (no < 10) ? tern(wf2[no * 128 + k]) : 0.f;
      Bf2[no * 128 + k] = (half_t)v;
    }
  }
}

// ============================================================
// Kernel 1: conv1 (3->16, K=27 pad 32) + bias + ReLU + maxpool2
// One workgroup = one image x 4 conv rows (8 waves, 16 px/wave).
// ============================================================
__global__ __launch_bounds__(256)
void conv1_kernel(const float* __restrict__ x, const float* __restrict__ b1,
                  const half_t* __restrict__ B1, half_t* __restrict__ h1) {
  int wg = blockIdx.x;
  int b  = wg >> 3;
  int y0 = (wg & 7) * 4;

  __shared__ float lds_x[3][6][34];                 // input halo tile (f32)
  __shared__ float lds_conv[128][16];               // conv out (px, ch)
  __shared__ __align__(16) half_t lds_B[16 * 32];   // weight B matrix

  int t = threadIdx.x;
  if (t < 64) ((v8h*)lds_B)[t] = ((const v8h*)B1)[t];
  for (int i = t; i < 3 * 6 * 34; i += 256) {
    int ic = i / 204, rem = i % 204, r = rem / 34, c = rem % 34;
    int iy = y0 + r - 1, ix = c - 1;
    float v = 0.f;
    if (iy >= 0 && iy < 32 && ix >= 0 && ix < 32)
      v = x[((b * 3 + ic) * 32 + iy) * 32 + ix];
    lds_x[ic][r][c] = v;
  }
  __syncthreads();

  int lane = t & 31, w = t >> 5;
  int mrow = lane & 15, hi = lane >> 4;
  int p = w * 16 + mrow;          // pixel within 4x32 region
  int prow = p >> 5, pcol = p & 31;

  v16h A;
#pragma unroll
  for (int h = 0; h < 16; ++h) {
    int k = (h < 8) ? (hi * 8 + h) : (16 + hi * 8 + (h - 8));
    float v = 0.f;
    if (k < 27) {
      int ic = k / 9, r = k % 9, ky = r / 3, kx = r % 3;
      v = lds_x[ic][prow + ky][pcol + kx];
    }
    A[h] = (half_t)v;
  }
  v16h Bf;
  {
    v8h c0 = *(const v8h*)&lds_B[mrow * 32 + hi * 8];
    v8h c1 = *(const v8h*)&lds_B[mrow * 32 + 16 + hi * 8];
    Bf = pack16(c0, c1);
  }
  v8f acc = {};
  acc = wmma_f16(A, Bf, acc);

#pragma unroll
  for (int r = 0; r < 8; ++r) {
    int m = r + hi * 8;                  // M row of the tile
    lds_conv[w * 16 + m][mrow] = acc[r]; // mrow == N == channel
  }
  __syncthreads();

  // pool 4x32 -> 2x16, bias + relu, write NHWC f16
  for (int o = t; o < 2 * 16 * 16; o += 256) {
    int ch = o & 15, px = (o >> 4) & 15, py = o >> 8;
    int p0 = (2 * py) * 32 + 2 * px;
    float v = fmaxf(fmaxf(lds_conv[p0][ch],      lds_conv[p0 + 1][ch]),
                    fmaxf(lds_conv[p0 + 32][ch], lds_conv[p0 + 33][ch]));
    v = fmaxf(v + b1[ch], 0.f);
    int gy = (y0 >> 1) + py;
    h1[(((size_t)b * 16 + gy) * 16 + px) * 16 + ch] = (half_t)v;
  }
}

// ============================================================
// Kernel 2: conv2 (16->32, K=144 pad 160) + bias + ReLU + pool2
// One workgroup = one image x 8 conv rows (8 waves, 16 px/wave,
// each wave does 2 N-tiles). h1 halo + weights staged in LDS;
// all fragment loads are 16B ds_load_b128.
// ============================================================
__global__ __launch_bounds__(256)
void conv2_kernel(const half_t* __restrict__ h1, const float* __restrict__ b2,
                  const half_t* __restrict__ B2, half_t* __restrict__ h2) {
  int wg = blockIdx.x;
  int b  = wg >> 1;
  int y0 = (wg & 1) * 8;

  __shared__ __align__(16) half_t lds_h1[10][18][16];  // halo tile, NHWC
  __shared__ __align__(16) half_t lds_B[32 * 160];     // weights
  __shared__ float lds_conv[128][32];                  // conv out (px, ch)

  int t = threadIdx.x;
  for (int i = t; i < (32 * 160) / 8; i += 256)
    ((v8h*)lds_B)[i] = ((const v8h*)B2)[i];
  for (int i = t; i < 10 * 18; i += 256) {
    int r = i / 18, c = i % 18;
    int iy = y0 + r - 1, ix = c - 1;
    v8h z = {}; v8h a0 = z, a1 = z;
    if (iy >= 0 && iy < 16 && ix >= 0 && ix < 16) {
      const v8h* src = (const v8h*)&h1[(((size_t)b * 16 + iy) * 16 + ix) * 16];
      a0 = src[0]; a1 = src[1];
    }
    v8h* dst = (v8h*)&lds_h1[r][c][0];
    dst[0] = a0; dst[1] = a1;
  }
  __syncthreads();

  int lane = t & 31, w = t >> 5;
  int mrow = lane & 15, hi = lane >> 4;
  int p = w * 16 + mrow;          // pixel within 8x16 region
  int prow = p >> 4, pcol = p & 15;

  v8f acc0 = {}, acc1 = {};
#pragma unroll
  for (int kt = 0; kt < 5; ++kt) {
    int k0 = kt * 32 + hi * 8;
    int k1 = k0 + 16;
    auto achunk = [&](int k) -> v8h {
      int s = k >> 4;           // spatial offset index 0..9
      v8h z = {};
      if (s >= 9) return z;     // K padding
      int c0 = k & 15;
      int ky = s / 3, kx = s % 3;
      return *(const v8h*)&lds_h1[prow + ky][pcol + kx][c0];
    };
    v16h A = pack16(achunk(k0), achunk(k1));
    int n0 = mrow, n1 = 16 + mrow;
    v16h Bt0 = pack16(*(const v8h*)&lds_B[n0 * 160 + k0],
                      *(const v8h*)&lds_B[n0 * 160 + k1]);
    v16h Bt1 = pack16(*(const v8h*)&lds_B[n1 * 160 + k0],
                      *(const v8h*)&lds_B[n1 * 160 + k1]);
    acc0 = wmma_f16(A, Bt0, acc0);
    acc1 = wmma_f16(A, Bt1, acc1);
  }

#pragma unroll
  for (int r = 0; r < 8; ++r) {
    int m = w * 16 + r + hi * 8;
    lds_conv[m][mrow]      = acc0[r];
    lds_conv[m][16 + mrow] = acc1[r];
  }
  __syncthreads();

  // pool 8x16 -> 4x8, bias + relu, write NHWC f16
  for (int o = t; o < 4 * 8 * 32; o += 256) {
    int ch = o & 31, px = (o >> 5) & 7, py = o >> 8;
    int p0 = (2 * py) * 16 + 2 * px;
    float v = fmaxf(fmaxf(lds_conv[p0][ch],      lds_conv[p0 + 1][ch]),
                    fmaxf(lds_conv[p0 + 16][ch], lds_conv[p0 + 17][ch]));
    v = fmaxf(v + b2[ch], 0.f);
    int gy = (y0 >> 1) + py;
    h2[(((size_t)b * 8 + gy) * 8 + px) * 32 + ch] = (half_t)v;
  }
}

// ============================================================
// Kernel 3: fc1  [B,2048] @ Bf1^T [2048,128] + bias + ReLU -> f16
// Block = 32 rows x 128 cols; wave w: M-tile w/4, N-tiles 2*(w&3)(+1).
// Fragments loaded straight from global as b128 (Bf1 lives in L2).
// ============================================================
__global__ __launch_bounds__(256)
void fc1_kernel(const half_t* __restrict__ h2f, const float* __restrict__ bf1,
                const half_t* __restrict__ Bf1, half_t* __restrict__ h3) {
  int t = threadIdx.x, lane = t & 31, w = t >> 5;
  int mrow = lane & 15, hi = lane >> 4;
  int mBase = blockIdx.x * 32 + (w >> 2) * 16;
  int row = mBase + mrow;
  int n0 = (w & 3) * 32 + mrow;   // column of acc0
  int n1 = n0 + 16;               // column of acc1

  const half_t* Arow = h2f + (size_t)row * 2048;
  const half_t* Bp0  = Bf1 + (size_t)n0 * 2048;
  const half_t* Bp1  = Bf1 + (size_t)n1 * 2048;

  v8f acc0 = {}, acc1 = {};
  for (int kt = 0; kt < 64; ++kt) {
    int k0 = kt * 32 + hi * 8;
    __builtin_prefetch(Arow + k0 + 64, 0, 0);  // global_prefetch next A
    v16h A  = pack16(*(const v8h*)(Arow + k0), *(const v8h*)(Arow + k0 + 16));
    v16h B0 = pack16(*(const v8h*)(Bp0 + k0),  *(const v8h*)(Bp0 + k0 + 16));
    v16h B1 = pack16(*(const v8h*)(Bp1 + k0),  *(const v8h*)(Bp1 + k0 + 16));
    acc0 = wmma_f16(A, B0, acc0);
    acc1 = wmma_f16(A, B1, acc1);
  }
  float bias0 = bf1[n0], bias1 = bf1[n1];
#pragma unroll
  for (int r = 0; r < 8; ++r) {
    int m = mBase + r + hi * 8;
    h3[(size_t)m * 128 + n0] = (half_t)fmaxf(acc0[r] + bias0, 0.f);
    h3[(size_t)m * 128 + n1] = (half_t)fmaxf(acc1[r] + bias1, 0.f);
  }
}

// ============================================================
// Kernel 4: fc2  [B,128] @ Bf2^T [128,10] + bias -> f32 out
// ============================================================
__global__ __launch_bounds__(256)
void fc2_kernel(const half_t* __restrict__ h3, const float* __restrict__ bf2,
                const half_t* __restrict__ Bf2, float* __restrict__ out) {
  int t = threadIdx.x, lane = t & 31, w = t >> 5;
  int mrow = lane & 15, hi = lane >> 4;
  int rowBase = blockIdx.x * 128 + w * 16;
  const half_t* Arow = h3 + (size_t)(rowBase + mrow) * 128;
  int n = mrow;
  const half_t* Bp = Bf2 + (size_t)n * 128;

  v8f acc = {};
#pragma unroll
  for (int kt = 0; kt < 4; ++kt) {
    int k0 = kt * 32 + hi * 8;
    v16h A = pack16(*(const v8h*)(Arow + k0), *(const v8h*)(Arow + k0 + 16));
    v16h B = pack16(*(const v8h*)(Bp + k0),   *(const v8h*)(Bp + k0 + 16));
    acc = wmma_f16(A, B, acc);
  }
  if (n < 10) {
    float bias = bf2[n];
#pragma unroll
    for (int r = 0; r < 8; ++r) {
      int m = rowBase + r + hi * 8;
      out[(size_t)m * 10 + n] = acc[r] + bias;
    }
  }
}

// ============================================================
extern "C" void kernel_launch(void* const* d_in, const int* in_sizes, int n_in,
                              void* d_out, int out_size, void* d_ws, size_t ws_size,
                              hipStream_t stream) {
  (void)n_in; (void)out_size; (void)ws_size;
  const float* x   = (const float*)d_in[0];
  const float* w1  = (const float*)d_in[1];
  const float* b1  = (const float*)d_in[2];
  const float* w2  = (const float*)d_in[3];
  const float* b2  = (const float*)d_in[4];
  const float* wf1 = (const float*)d_in[5];
  const float* bf1 = (const float*)d_in[6];
  const float* wf2 = (const float*)d_in[7];
  const float* bf2 = (const float*)d_in[8];
  float* out = (float*)d_out;

  int B = in_sizes[0] / (3 * 32 * 32);   // 4096 for the reference

  char* ws = (char*)d_ws;
  half_t* B1w  = (half_t*)(ws + OFF_B1);
  half_t* B2w  = (half_t*)(ws + OFF_B2);
  half_t* Bf1w = (half_t*)(ws + OFF_BF1);
  half_t* Bf2w = (half_t*)(ws + OFF_BF2);
  half_t* h1   = (half_t*)(ws + OFF_H1);
  half_t* h2   = (half_t*)(ws + OFF_H2);
  half_t* h3   = (half_t*)(ws + OFF_H3);

  tern_pack_kernel<<<4, 256, 0, stream>>>(w1, w2, wf1, wf2, B1w, B2w, Bf1w, Bf2w);
  conv1_kernel<<<B * 8, 256, 0, stream>>>(x, b1, B1w, h1);
  conv2_kernel<<<B * 2, 256, 0, stream>>>(h1, b2, B2w, h2);
  fc1_kernel<<<B / 32, 256, 0, stream>>>(h2, bf1, Bf1w, h3);
  fc2_kernel<<<B / 128, 256, 0, stream>>>(h3, bf2, Bf2w, out);
}
